// WindowAttention_37632503447910
// MI455X (gfx1250) — compile-verified, matrix operand order checked by reading, not run on hardware
//
#include <hip/hip_runtime.h>
#include <hip/hip_bf16.h>
#include <cstdint>

typedef _Float16 v8h  __attribute__((ext_vector_type(8)));
typedef _Float16 v16h __attribute__((ext_vector_type(16)));
typedef float    v8f  __attribute__((ext_vector_type(8)));
typedef float    v4f  __attribute__((ext_vector_type(4)));

#define B_WIN   4096
#define NTOK    49
#define DIMC    512
#define HEADS   16
#define HD      32
#define MTOT    (B_WIN * NTOK)       // 200704
#define SCALE_F 0.1767766952966369f  // 32^-0.5

// ---------- CDNA5 helpers ----------

static __device__ __forceinline__ unsigned lds_off(const void* p) {
  // generic LDS pointer: low 32 bits are the LDS byte offset (flat aperture rule)
  return (unsigned)(unsigned long long)p;
}

static __device__ __forceinline__ void async_b128(unsigned ldsoff, const void* gptr) {
  // GLOBAL_LOAD_ASYNC_TO_LDS_B128 (ASYNCcnt-tracked DMA into LDS)
  asm volatile("global_load_async_to_lds_b128 %0, %1, off"
               :: "v"(ldsoff), "v"(gptr) : "memory");
}

static __device__ __forceinline__ void wait_async0() {
  asm volatile("s_wait_asynccnt 0" ::: "memory");
}
static __device__ __forceinline__ void wait_async2() {
  // async loads complete in order: <=2 outstanding => previous tile has landed
  asm volatile("s_wait_asynccnt 2" ::: "memory");
}
static __device__ __forceinline__ void wait_ds0() {
  asm volatile("s_wait_dscnt 0" ::: "memory");
}

// Load one 16x32 f16 A/B fragment from an LDS row (per-lane M/N index already
// folded into rowPtr). Lane l (g=l>>4) holds K=[8g..8g+7] and [16+8g..16+8g+7]
// per ISA 7.12.2. Two ds_load_b128 per fragment.
static __device__ __forceinline__ v16h ldfrag(const _Float16* rowPtr, int g) {
  v8h lo = *(const v8h*)(rowPtr + 8 * g);
  v8h hi = *(const v8h*)(rowPtr + 16 + 8 * g);
  return __builtin_shufflevector(lo, hi, 0,1,2,3,4,5,6,7,8,9,10,11,12,13,14,15);
}

static __device__ __forceinline__ v8f wmma16(v16h a, v16h b, v8f c) {
  return __builtin_amdgcn_wmma_f32_16x16x32_f16(false, a, false, b, (short)0, c,
                                                false, false);
}

// ---------- Kernel A: weight fp32 -> f16 (fold SCALE into Wq rows) ----------

__global__ __launch_bounds__(256) void wa_convert(const float* __restrict__ qkv_w,
                                                  const float* __restrict__ proj_w,
                                                  _Float16* __restrict__ wq,
                                                  _Float16* __restrict__ wp) {
  int idx = blockIdx.x * 256 + threadIdx.x;          // 1,048,576 total
  if (idx < 3 * DIMC * DIMC) {                       // 786432: qkv_w
    float v = qkv_w[idx];
    if (idx < DIMC * DIMC) v *= SCALE_F;             // q rows pre-scaled
    wq[idx] = (_Float16)v;
  } else {
    int j = idx - 3 * DIMC * DIMC;
    wp[j] = (_Float16)proj_w[j];
  }
}

// ---------- Kernel B: QKV GEMM ----------
// 128-row A stripe resident in 128KB LDS (x read exactly once); weight tiles
// double-buffered through async DMA so the next tile's transfer overlaps the
// current tile's 8 WMMAs. out[m,c] = x[m,:].Wqkv[c,:] + b[c], scattered f16
// into q/k/v laid out [B,H,N,hd].

__global__ __launch_bounds__(256) void wa_qkv_gemm(const float* __restrict__ x,
                                                   const _Float16* __restrict__ wq,
                                                   const float* __restrict__ qkv_b,
                                                   _Float16* __restrict__ qws,
                                                   _Float16* __restrict__ kws,
                                                   _Float16* __restrict__ vws) {
  __shared__ __align__(16) _Float16 As[128 * 512];     // 128 KB
  __shared__ __align__(16) _Float16 Bs[2][128 * 32];   // 2 x 8 KB double buffer
  const int tid = threadIdx.x, lane = tid & 31, wid = tid >> 5;
  const int wm = wid >> 2, wn = wid & 3;               // 2x4 waves -> 64x32/wave
  const int lm = lane & 15, g = lane >> 4;
  const long gm0 = (long)blockIdx.x * 128;

  // per-thread B-tile chunk geometry (2 x 16B chunks per thread per tile)
  const int q0 = tid * 2;
  const int n0 = q0 >> 2,      c0 = (q0 & 3) * 8;
  const int n1 = (q0 + 1) >> 2, c1 = ((q0 + 1) & 3) * 8;
  auto issue_bs = [&](int t, int buf) {
    int cn0 = (t >> 4) * 128, k0 = (t & 15) * 32;
    async_b128(lds_off(&Bs[buf][n0 * 32 + c0]), wq + (long)(cn0 + n0) * 512 + k0 + c0);
    async_b128(lds_off(&Bs[buf][n1 * 32 + c1]), wq + (long)(cn0 + n1) * 512 + k0 + c1);
  };

  issue_bs(0, 0);  // prefetch first weight tile while we stage A

  // Stage A once: fp32 -> f16, linear layout matches global (stride 512)
  const float* xs = x + gm0 * 512;
  for (int c = tid; c < (128 * 512) / 16; c += 256) {
    int e = c * 16;
    v4f f0 = *(const v4f*)(xs + e + 0);
    v4f f1 = *(const v4f*)(xs + e + 4);
    v4f f2 = *(const v4f*)(xs + e + 8);
    v4f f3 = *(const v4f*)(xs + e + 12);
    v8h h0, h1;
#pragma unroll
    for (int j = 0; j < 4; ++j) {
      h0[j] = (_Float16)f0[j]; h0[4 + j] = (_Float16)f1[j];
      h1[j] = (_Float16)f2[j]; h1[4 + j] = (_Float16)f3[j];
    }
    *(v8h*)(As + e) = h0;
    *(v8h*)(As + e + 8) = h1;
  }

  v8f acc[4][2];
  const v8f zf = {};
  for (int t = 0; t < 12 * 16; ++t) {                  // (nb, k-step) pipeline
    const int cur = t & 1;
    if ((t & 15) == 0) {
#pragma unroll
      for (int mi = 0; mi < 4; ++mi)
#pragma unroll
        for (int ni = 0; ni < 2; ++ni) acc[mi][ni] = zf;
    }
    if (t + 1 < 12 * 16) { issue_bs(t + 1, cur ^ 1); wait_async2(); }
    else                 { wait_async0(); }
    __syncthreads();                                   // tile t visible to all

    const int k0 = (t & 15) * 32;
    v16h af[4], bf[2];
#pragma unroll
    for (int mi = 0; mi < 4; ++mi)
      af[mi] = ldfrag(As + (wm * 64 + mi * 16 + lm) * 512 + k0, g);
#pragma unroll
    for (int ni = 0; ni < 2; ++ni)
      bf[ni] = ldfrag(&Bs[cur][(wn * 32 + ni * 16 + lm) * 32], g);
#pragma unroll
    for (int mi = 0; mi < 4; ++mi)
#pragma unroll
      for (int ni = 0; ni < 2; ++ni)
        acc[mi][ni] = wmma16(af[mi], bf[ni], acc[mi][ni]);
    __syncthreads();                                   // reads done before reuse

    if ((t & 15) == 15) {                              // epilogue for this nb
      const int cn0 = (t >> 4) * 128;
#pragma unroll
      for (int ni = 0; ni < 2; ++ni) {
        int cc = cn0 + wn * 32 + ni * 16 + lm;
        float bias = qkv_b[cc];
        int part = cc >> 9, rem = cc & 511, hh = rem >> 5, dd = rem & 31;
        if (part == 0) bias *= SCALE_F;
        _Float16* dst = (part == 0) ? qws : ((part == 1) ? kws : vws);
#pragma unroll
        for (int mi = 0; mi < 4; ++mi)
#pragma unroll
          for (int r = 0; r < 8; ++r) {
            int gm = (int)gm0 + wm * 64 + mi * 16 + r + 8 * g;
            int bw = gm / 49, ii = gm - bw * 49;
            dst[(((long)bw * 16 + hh) * 49 + ii) * 32 + dd] =
                (_Float16)(acc[mi][ni][r] + bias);
          }
      }
    }
  }
}

// ---------- Kernel C: per-(window,head) attention, 1 wave per task ----------

__global__ __launch_bounds__(32) void wa_attn(const _Float16* __restrict__ qws,
                                              const _Float16* __restrict__ kws,
                                              const _Float16* __restrict__ vws,
                                              const float* __restrict__ mask,
                                              const float* __restrict__ table,
                                              _Float16* __restrict__ attn_ws) {
  __shared__ __align__(16) _Float16 Qs[64 * 32];     // 49 rows valid, rest 0
  __shared__ __align__(16) _Float16 Ks[64 * 32];
  __shared__ __align__(16) _Float16 Vt[32 * 64];     // V transposed (d, token)
  __shared__ __align__(16) float    S [64 * 64];     // fp32 scores
  __shared__ __align__(16) _Float16 Ps[64 * 64];     // softmax probs, f16
  const int lane = threadIdx.x;
  const int bh = blockIdx.x, b = bh >> 4, h = bh & 15;
  const long base = (long)bh * NTOK * HD;            // contiguous 49x32 f16
  const int lm = lane & 15, g = lane >> 4;

  // zero pads
  v8h z = {};
  for (int i = lane; i < (64 * 32) / 8; i += 32) {
    ((v8h*)Qs)[i] = z; ((v8h*)Ks)[i] = z; ((v8h*)Vt)[i] = z;
  }
  for (int i = lane; i < (64 * 64) / 8; i += 32) ((v8h*)Ps)[i] = z;
  wait_ds0();
  __syncthreads();

  // async-stage Q,K (49*32 f16 = 196 x 16B, contiguous, row stride 64B == LDS)
  for (int c = lane; c < 196; c += 32) {
    async_b128(lds_off(Qs + c * 8), qws + base + c * 8);
    async_b128(lds_off(Ks + c * 8), kws + base + c * 8);
  }
  // stage V transposed: wide global loads, scalar transposed LDS stores
  for (int c = lane; c < 196; c += 32) {
    int e = c * 8;                                   // 8 consecutive d, same row
    int j = e >> 5, d = e & 31;
    v8h vv = *(const v8h*)(vws + base + e);
#pragma unroll
    for (int u = 0; u < 8; ++u) Vt[(d + u) * 64 + j] = vv[u];
  }
  wait_async0();
  __syncthreads();

  // S = Q (49x32, pre-scaled) @ K^T -> 4x4 WMMA tiles
#pragma unroll
  for (int mi = 0; mi < 4; ++mi) {
    v16h aq = ldfrag(Qs + (mi * 16 + lm) * 32, g);
#pragma unroll
    for (int ni = 0; ni < 4; ++ni) {
      v16h bk = ldfrag(Ks + (ni * 16 + lm) * 32, g);
      v8f c = {};
      c = wmma16(aq, bk, c);
#pragma unroll
      for (int r = 0; r < 8; ++r)
        S[(mi * 16 + r + 8 * g) * 64 + ni * 16 + lm] = c[r];
    }
  }
  __syncthreads();

  // softmax rows (bias via on-the-fly relative-position index; mask[b%64])
  const float* mrow = mask + (long)(b & 63) * NTOK * NTOK;
  for (int row = lane; row < NTOK; row += 32) {
    int ih = row / 7, iw = row - ih * 7;
    float mx = -3.0e38f;
    for (int c2 = 0; c2 < NTOK; ++c2) {
      int jh = c2 / 7, jw = c2 - jh * 7;
      int rel = (ih - jh + 6) * 13 + (iw - jw + 6);
      float v = S[row * 64 + c2] + table[rel * HEADS + h] + mrow[row * NTOK + c2];
      S[row * 64 + c2] = v;
      mx = fmaxf(mx, v);
    }
    float sum = 0.f;
    for (int c2 = 0; c2 < NTOK; ++c2) {
      float e = __expf(S[row * 64 + c2] - mx);
      S[row * 64 + c2] = e;
      sum += e;
    }
    float inv = 1.0f / sum;
    for (int c2 = 0; c2 < NTOK; ++c2)
      Ps[row * 64 + c2] = (_Float16)(S[row * 64 + c2] * inv);
  }
  __syncthreads();

  // out = P (49x64) @ V (64x32): K=64 -> 2 k-steps, 4x2 tiles
#pragma unroll
  for (int mi = 0; mi < 4; ++mi)
#pragma unroll
    for (int ni = 0; ni < 2; ++ni) {
      v8f c = {};
#pragma unroll
      for (int kk = 0; kk < 64; kk += 32) {
        v16h ap = ldfrag(Ps + (mi * 16 + lm) * 64 + kk, g);
        v16h bv = ldfrag(Vt + (ni * 16 + lm) * 64 + kk, g);
        c = wmma16(ap, bv, c);
      }
#pragma unroll
      for (int r = 0; r < 8; ++r) {
        int ti = mi * 16 + r + 8 * g;
        if (ti < NTOK)
          attn_ws[((long)b * NTOK + ti) * DIMC + h * HD + ni * 16 + lm] =
              (_Float16)c[r];
      }
    }
}

// ---------- Kernel D: projection GEMM (all-async staging, fp32 out) ----------

__global__ __launch_bounds__(256) void wa_proj_gemm(const _Float16* __restrict__ attn,
                                                    const _Float16* __restrict__ wp,
                                                    const float* __restrict__ proj_b,
                                                    float* __restrict__ out) {
  __shared__ __align__(16) _Float16 As[128 * 512];
  __shared__ __align__(16) _Float16 Bs[2][128 * 32];
  const int tid = threadIdx.x, lane = tid & 31, wid = tid >> 5;
  const int wm = wid >> 2, wn = wid & 3;
  const int lm = lane & 15, g = lane >> 4;
  const long gm0 = (long)blockIdx.x * 128;

  const int q0 = tid * 2;
  const int n0 = q0 >> 2,       c0 = (q0 & 3) * 8;
  const int n1 = (q0 + 1) >> 2, c1 = ((q0 + 1) & 3) * 8;
  auto issue_bs = [&](int t, int buf) {
    int cn0 = (t >> 4) * 128, k0 = (t & 15) * 32;
    async_b128(lds_off(&Bs[buf][n0 * 32 + c0]), wp + (long)(cn0 + n0) * 512 + k0 + c0);
    async_b128(lds_off(&Bs[buf][n1 * 32 + c1]), wp + (long)(cn0 + n1) * 512 + k0 + c1);
  };

  // async-stage the whole 128KB A stripe (contiguous f16), then first B tile;
  // in-order completion means waiting on the B tile also covers the A stripe.
  const _Float16* asrc = attn + gm0 * 512;
  for (int c = tid; c < (128 * 512) / 8; c += 256)
    async_b128(lds_off(As + c * 8), asrc + c * 8);
  issue_bs(0, 0);

  v8f acc[4][2];
  const v8f zf = {};
  for (int t = 0; t < 4 * 16; ++t) {
    const int cur = t & 1;
    if ((t & 15) == 0) {
#pragma unroll
      for (int mi = 0; mi < 4; ++mi)
#pragma unroll
        for (int ni = 0; ni < 2; ++ni) acc[mi][ni] = zf;
    }
    if (t + 1 < 4 * 16) { issue_bs(t + 1, cur ^ 1); wait_async2(); }
    else                { wait_async0(); }
    __syncthreads();

    const int k0 = (t & 15) * 32;
    v16h af[4], bf[2];
#pragma unroll
    for (int mi = 0; mi < 4; ++mi)
      af[mi] = ldfrag(As + (wm * 64 + mi * 16 + lm) * 512 + k0, g);
#pragma unroll
    for (int ni = 0; ni < 2; ++ni)
      bf[ni] = ldfrag(&Bs[cur][(wn * 32 + ni * 16 + lm) * 32], g);
#pragma unroll
    for (int mi = 0; mi < 4; ++mi)
#pragma unroll
      for (int ni = 0; ni < 2; ++ni)
        acc[mi][ni] = wmma16(af[mi], bf[ni], acc[mi][ni]);
    __syncthreads();

    if ((t & 15) == 15) {
      const int cn0 = (t >> 4) * 128;
#pragma unroll
      for (int ni = 0; ni < 2; ++ni) {
        int cc = cn0 + wn * 32 + ni * 16 + lm;
        float bias = proj_b[cc];
#pragma unroll
        for (int mi = 0; mi < 4; ++mi)
#pragma unroll
          for (int r = 0; r < 8; ++r) {
            long gm = gm0 + wm * 64 + mi * 16 + r + 8 * g;
            out[gm * 512 + cc] = acc[mi][ni][r] + bias;
          }
      }
    }
  }
}

// ---------- host ----------

extern "C" void kernel_launch(void* const* d_in, const int* in_sizes, int n_in,
                              void* d_out, int out_size, void* d_ws, size_t ws_size,
                              hipStream_t stream) {
  const float* x      = (const float*)d_in[0];
  const float* mask   = (const float*)d_in[1];
  const float* qkv_w  = (const float*)d_in[2];
  const float* qkv_b  = (const float*)d_in[3];
  const float* table  = (const float*)d_in[4];
  const float* proj_w = (const float*)d_in[5];
  const float* proj_b = (const float*)d_in[6];
  float* out = (float*)d_out;

  char* ws = (char*)d_ws;
  size_t o = 0;
  auto take = [&](size_t bytes) { size_t r = o; o += (bytes + 255) & ~(size_t)255; return r; };
  _Float16* wq  = (_Float16*)(ws + take((size_t)3 * DIMC * DIMC * 2));
  _Float16* wp  = (_Float16*)(ws + take((size_t)DIMC * DIMC * 2));
  const size_t qkv_elems = (size_t)B_WIN * HEADS * NTOK * HD;
  _Float16* qws = (_Float16*)(ws + take(qkv_elems * 2));
  _Float16* kws = (_Float16*)(ws + take(qkv_elems * 2));
  _Float16* vws = (_Float16*)(ws + take(qkv_elems * 2));
  _Float16* aws = (_Float16*)(ws + take((size_t)MTOT * DIMC * 2));

  wa_convert<<<dim3((3 * DIMC * DIMC + DIMC * DIMC) / 256), dim3(256), 0, stream>>>(
      qkv_w, proj_w, wq, wp);
  wa_qkv_gemm<<<dim3(MTOT / 128), dim3(256), 0, stream>>>(x, wq, qkv_b, qws, kws, vws);
  wa_attn<<<dim3(B_WIN * HEADS), dim3(32), 0, stream>>>(qws, kws, vws, mask, table, aws);
  wa_proj_gemm<<<dim3(MTOT / 128), dim3(256), 0, stream>>>(aws, wp, proj_b, out);
}
